// IndexStaged_17772574671113
// MI455X (gfx1250) — matrix operand order repaired and checked by reading, not compiled, not run on hardware
//
#include <hip/hip_runtime.h>
#include <hip/hip_bf16.h>

typedef __attribute__((ext_vector_type(16))) _Float16 v16h;
typedef __attribute__((ext_vector_type(8)))  float    v8f;
typedef __attribute__((ext_vector_type(4)))  int      v4i;

#define DEG 32
#define N_ITERS 6
#define NFULL 131072
#define NSAMP 32768
#define DFULL 512
#define BQ 32
#define KTOP 16
#define INFF  (__builtin_inff())
#define NANF  (__uint_as_float(0x7fc00000u))

#define GLOBAL_AS __attribute__((address_space(1)))
#define LDS_AS    __attribute__((address_space(3)))

typedef GLOBAL_AS v4i* gv4i_p;
typedef LDS_AS    v4i* lv4i_p;

// ---------------------------------------------------------------------------
// Async global->LDS 16B copy (CDNA5 GLOBAL_LOAD_ASYNC_TO_LDS_B128, ASYNCcnt).
// Guarded: falls back to plain vector loads if the builtin is absent.
// ---------------------------------------------------------------------------
__device__ __forceinline__ void async_copy16(const void* gsrc, void* ldst) {
#if defined(__gfx1250__) && __has_builtin(__builtin_amdgcn_global_load_async_to_lds_b128)
    __builtin_amdgcn_global_load_async_to_lds_b128(
        (gv4i_p)gsrc, (lv4i_p)ldst, 0, 0);
#else
    *(float4*)ldst = *(const float4*)gsrc;
#endif
}

__device__ __forceinline__ void async_wait0() {
#if defined(__gfx1250__) && __has_builtin(__builtin_amdgcn_global_load_async_to_lds_b128)
#if __has_builtin(__builtin_amdgcn_s_wait_asynccnt)
    __builtin_amdgcn_s_wait_asynccnt(0);
#else
    asm volatile("s_wait_asynccnt 0" ::: "memory");
#endif
#endif
}

// ---------------------------------------------------------------------------
// Kernel 1: q = query(32x512) @ VT^T(512x512) via v_wmma_f32_16x16x32_f16.
// One wave per 16x16 output tile. grid = (32 col tiles, 2 row tiles).
// A fragment (16x32 f16): lane<16 holds K={e, e+8}, lane>=16 holds K={e+8, e+16}
// B fragment (32x16 f16): lane gives N=lane%16, element e -> K = (e<8 ? e : e+8) + (lane>=16 ? 8 : 0)
// C/D (16x16 f32, 8 VGPR): reg r -> M = r + (lane>=16 ? 8 : 0), N = lane%16
// ---------------------------------------------------------------------------
__global__ __launch_bounds__(32) void project_kernel(const float* __restrict__ query,
                                                     const float* __restrict__ VT,
                                                     float* __restrict__ qout) {
    const int lane = threadIdx.x;
    const int tile_i = blockIdx.x;   // output column tile (0..31)
    const int tile_b = blockIdx.y;   // output row tile (0..1)
    const int mn = lane & 15;
    const bool hi = lane >= 16;

    v8f c = {};
    const float* qrow = query + (size_t)(tile_b * 16 + mn) * DFULL;
    const float* vrow = VT    + (size_t)(tile_i * 16 + mn) * DFULL;

#pragma unroll
    for (int kb = 0; kb < DFULL; kb += 32) {
        v16h a, b;
#pragma unroll
        for (int e = 0; e < 16; ++e) {
            int ka = (e < 8 ? e : e + 8) + (hi ? 8 : 0);   // A-fragment K index
            int kbb = (e < 8 ? e : e + 8) + (hi ? 8 : 0);  // B-fragment K index
            a[e] = (_Float16)qrow[kb + ka];
            b[e] = (_Float16)vrow[kb + kbb];
        }
        c = __builtin_amdgcn_wmma_f32_16x16x32_f16(false, a, false, b,
                                                   (short)0, c, false, false);
    }
#pragma unroll
    for (int r = 0; r < 8; ++r) {
        int row = tile_b * 16 + r + (hi ? 8 : 0);
        int col = tile_i * 16 + mn;
        qout[(size_t)row * DFULL + col] = c[r];
    }
}

// ---------------------------------------------------------------------------
// Beam-search kernel: one workgroup (256 threads = 8 waves) per query.
// MODE 1: stage1  EF=128, D=128, sampled domain, entries = {0}
// MODE 2: stage2  EF=32,  D=512, sampled domain, entries = pi1[:, :32], persist visited
// MODE 3: stage3  EF=32,  D=512, full domain, entries = sampled[pi2[:, :16]] edge-padded,
//                 visited seeded by scattering stage2 bitmap through sampled_ids
// ---------------------------------------------------------------------------
template<int EF, int DIMS, int NVIS_BITS, int MODE>
__global__ __launch_bounds__(256) void beam_kernel(
        const float* __restrict__ q,          // BQ x 512 projected queries (f32)
        const float* __restrict__ xb,         // NFULL x 512
        const int*   __restrict__ graph,      // (Ns|N) x DEG
        const int*   __restrict__ sampled,    // NSAMP
        const int*   __restrict__ entries_in, // stage2: pi1 (Bx128); stage3: pi2 (Bx32)
        int*         __restrict__ pi_out,     // stage1: Bx128; stage2: Bx32
        unsigned*    __restrict__ vis2_out,   // stage2 only: Bx1024 words
        const unsigned* __restrict__ vis2_in, // stage3 only: Bx1024 words
        unsigned long long* __restrict__ vis_counter,
        float*       __restrict__ dout)       // stage3 only
{
    constexpr int NC     = EF * DEG;          // neighbor candidates per iter
    constexpr int M      = EF + NC;           // merged candidate count
    constexpr int NWORDS = NVIS_BITS / 32;
    constexpr bool USE_SAMPLED = (MODE != 3);

    __shared__ unsigned vis[NWORDS];
    __shared__ float qs[DIMS];
    __shared__ int   pi[EF];
    __shared__ float pd[EF];
    __shared__ int   ai[M];
    __shared__ float ad[M];
    __shared__ float red_d[8];
    __shared__ int   red_i[8];
    __shared__ unsigned long long blk_cnt;

    const int tid  = threadIdx.x;
    const int wave = tid >> 5;
    const int lane = tid & 31;
    const int b    = blockIdx.x;

    // ---- init: clear visited; wave0 stages query into LDS via async copy ----
    for (int w = tid; w < NWORDS; w += 256) vis[w] = 0u;
    if (tid == 0) blk_cnt = 0ull;
    if (wave == 0) {
        const char* gsrc = (const char*)(q + (size_t)b * DFULL);
        char* ldst = (char*)qs;
        for (int t = lane; t < (DIMS * 4) / 16; t += 32)
            async_copy16(gsrc + t * 16, ldst + t * 16);
        async_wait0();
    }
    __syncthreads();

    // ---- stage3: scatter stage2 visited bitmap through sampled_ids ----
    if (MODE == 3) {
        for (int w = tid; w < NSAMP / 32; w += 256) {
            unsigned m = vis2_in[(size_t)b * (NSAMP / 32) + w];
            while (m) {
                int bit = __ffs(m) - 1;
                m &= m - 1;
                int id = sampled[w * 32 + bit];
                atomicOr(&vis[id >> 5], 1u << (id & 31));
            }
        }
        __syncthreads();
    }

    // ---- entries ----
    for (int j = tid; j < EF; j += 256) {
        int e;
        if (MODE == 1)      e = 0;
        else if (MODE == 2) e = entries_in[(size_t)b * 128 + j];
        else                e = sampled[entries_in[(size_t)b * 32 + (j < KTOP ? j : KTOP - 1)]];
        ai[j] = e;
    }
    __syncthreads();

    // entry distances (no visited mask, matches reference d0)
    for (int cc = wave; cc < EF; cc += 8) {
        int nb = ai[cc];
        const float* v = xb + (size_t)(USE_SAMPLED ? sampled[nb] : nb) * DFULL;
        float acc = 0.f;
        for (int d = lane; d < DIMS; d += 32) { float t = v[d] - qs[d]; acc += t * t; }
#pragma unroll
        for (int off = 16; off > 0; off >>= 1) acc += __shfl_xor(acc, off, 32);
        if (lane == 0) ad[cc] = acc;
    }
    __syncthreads();
    // mark entries visited
    for (int j = tid; j < EF; j += 256) {
        int nb = ai[j];
        atomicOr(&vis[nb >> 5], 1u << (nb & 31));
    }
    for (int j = tid; j < EF; j += 256) { pi[j] = ai[j]; pd[j] = ad[j]; }
    __syncthreads();
    // dedup entries: later duplicates -> inf
    for (int j = tid; j < EF; j += 256) {
        int me = pi[j];
        for (int j2 = 0; j2 < j; ++j2)
            if (pi[j2] == me) { pd[j] = INFF; break; }
    }
    __syncthreads();

    // ---- main loop ----
    for (int it = 0; it < N_ITERS; ++it) {
        // build merged candidate list: [beam | neighbors]
        for (int j = tid; j < EF; j += 256) { ai[j] = pi[j]; ad[j] = pd[j]; }
        for (int j = tid; j < NC; j += 256) {
            int src = pi[j / DEG];
            ai[EF + j] = graph[(size_t)src * DEG + (j % DEG)];
        }
        __syncthreads();

        // distances: one wave per candidate; visited -> inf (pre-mark state)
        for (int cc = wave; cc < NC; cc += 8) {
            int nb = ai[EF + cc];
            const float* v = xb + (size_t)(USE_SAMPLED ? sampled[nb] : nb) * DFULL;
            int ncc = cc + 8;
            if (ncc < NC) {
                int nb2 = ai[EF + ncc];
                const float* v2 = xb + (size_t)(USE_SAMPLED ? sampled[nb2] : nb2) * DFULL;
                __builtin_prefetch(v2 + lane * 4, 0, 1);
            }
            float acc = 0.f;
            for (int d = lane; d < DIMS; d += 32) { float t = v[d] - qs[d]; acc += t * t; }
#pragma unroll
            for (int off = 16; off > 0; off >>= 1) acc += __shfl_xor(acc, off, 32);
            if (lane == 0) {
                bool seen = (vis[nb >> 5] >> (nb & 31)) & 1u;
                ad[EF + cc] = seen ? INFF : acc;
            }
        }
        __syncthreads();
        // mark all neighbors visited
        for (int j = tid; j < NC; j += 256) {
            int nb = ai[EF + j];
            atomicOr(&vis[nb >> 5], 1u << (nb & 31));
        }
        __syncthreads();

        // stable top-EF by min distance (ties -> lower position, like lax.top_k)
        // per-wave shfl argmin + 8-way merge: 2 barriers per selection
        for (int s = 0; s < EF; ++s) {
            float bd = INFF; int bi = 0x7fffffff;
            for (int idx = tid; idx < M; idx += 256) {
                float d = ad[idx];
                if (d < bd || (d == bd && idx < bi)) { bd = d; bi = idx; }
            }
#pragma unroll
            for (int off = 16; off > 0; off >>= 1) {
                float od = __shfl_xor(bd, off, 32);
                int   oi = __shfl_xor(bi, off, 32);
                if (od < bd || (od == bd && oi < bi)) { bd = od; bi = oi; }
            }
            if (lane == 0) { red_d[wave] = bd; red_i[wave] = bi; }
            __syncthreads();
            if (tid == 0) {
                float fb = red_d[0]; int fi = red_i[0];
#pragma unroll
                for (int w2 = 1; w2 < 8; ++w2) {
                    if (red_d[w2] < fb || (red_d[w2] == fb && red_i[w2] < fi)) {
                        fb = red_d[w2]; fi = red_i[w2];
                    }
                }
                pi[s] = ai[fi];
                pd[s] = fb;
                ad[fi] = NANF;   // NaN never re-selected (all compares false)
            }
            __syncthreads();
        }
        // dedup selected beam
        for (int j = tid; j < EF; j += 256) {
            int me = pi[j];
            for (int j2 = 0; j2 < j; ++j2)
                if (pi[j2] == me) { pd[j] = INFF; break; }
        }
        __syncthreads();
    }

    // ---- outputs ----
    if (MODE == 1) {
        for (int j = tid; j < EF; j += 256) pi_out[(size_t)b * 128 + j] = pi[j];
    } else if (MODE == 2) {
        for (int j = tid; j < EF; j += 256) pi_out[(size_t)b * 32 + j] = pi[j];
        for (int w = tid; w < NWORDS; w += 256) vis2_out[(size_t)b * NWORDS + w] = vis[w];
    } else {
        for (int j = tid; j < KTOP; j += 256) {
            dout[(size_t)b * KTOP + j]               = (float)pi[j];
            dout[BQ * KTOP + (size_t)b * KTOP + j]   = pd[j];
        }
    }

    // visited count
    unsigned long long local = 0;
    for (int w = tid; w < NWORDS; w += 256) local += (unsigned long long)__popc(vis[w]);
    atomicAdd(&blk_cnt, local);
    __syncthreads();
    if (tid == 0) atomicAdd(vis_counter, blk_cnt);
}

__global__ void init_kernel(unsigned long long* cnt) {
    if (threadIdx.x < 3) cnt[threadIdx.x] = 0ull;
}

__global__ void finalize_kernel(const unsigned long long* __restrict__ cnt,
                                float* __restrict__ dout) {
    if (threadIdx.x < 3) dout[2 * BQ * KTOP + threadIdx.x] = (float)cnt[threadIdx.x];
}

extern "C" void kernel_launch(void* const* d_in, const int* in_sizes, int n_in,
                              void* d_out, int out_size, void* d_ws, size_t ws_size,
                              hipStream_t stream) {
    const float* query         = (const float*)d_in[0];
    const float* VT            = (const float*)d_in[1];
    const float* xb            = (const float*)d_in[2];
    const int*   graph_full    = (const int*)d_in[3];
    const int*   graph_sampled = (const int*)d_in[4];
    const int*   sampled       = (const int*)d_in[5];
    float* dout = (float*)d_out;

    char* ws = (char*)d_ws;
    float*    qproj = (float*)ws;                                   // 32*512*4   = 65536
    int*      pi1   = (int*)(ws + 65536);                           // 32*128*4   = 16384
    int*      pi2   = (int*)(ws + 65536 + 16384);                   // 32*32*4    = 4096
    unsigned* vis2  = (unsigned*)(ws + 65536 + 16384 + 4096);       // 32*1024*4  = 131072
    unsigned long long* cnt =
        (unsigned long long*)(ws + 65536 + 16384 + 4096 + 131072);  // 3*8

    init_kernel<<<1, 32, 0, stream>>>(cnt);
    project_kernel<<<dim3(32, 2), 32, 0, stream>>>(query, VT, qproj);

    // stage 1: EF=128, D=128, sampled domain
    beam_kernel<128, 128, NSAMP, 1><<<BQ, 256, 0, stream>>>(
        qproj, xb, graph_sampled, sampled, nullptr, pi1, nullptr, nullptr, cnt + 0, nullptr);
    // stage 2: EF=32, D=512, sampled domain, persist visited
    beam_kernel<32, 512, NSAMP, 2><<<BQ, 256, 0, stream>>>(
        qproj, xb, graph_sampled, sampled, pi1, pi2, vis2, nullptr, cnt + 1, nullptr);
    // stage 3: EF=32, D=512, full domain, seed visited from stage2
    beam_kernel<32, 512, NFULL, 3><<<BQ, 256, 0, stream>>>(
        qproj, xb, graph_full, sampled, pi2, nullptr, nullptr, vis2, cnt + 2, dout);

    finalize_kernel<<<1, 32, 0, stream>>>(cnt, dout);
}
